// Net3_88106959110558
// MI455X (gfx1250) — compile-verified, matrix operand order checked by reading, not run on hardware
//
#include <hip/hip_runtime.h>
#include <cstdint>
#include <cstddef>

typedef __attribute__((ext_vector_type(8)))  float  v8f;
typedef __attribute__((ext_vector_type(16))) __bf16 v16bf;

// ---------------------------------------------------------------- utilities
__device__ __forceinline__ __bf16 f2bf(float f) {
    return (__bf16)f;   // native v_cvt (RNE) on gfx1250
}

__device__ __forceinline__ void atomicMaxF(float* addr, float v) {
    if (v >= 0.0f) atomicMax((int*)addr, __float_as_int(v));
    else           atomicMin((unsigned int*)addr, __float_as_uint(v));
}

__global__ void k_fill(unsigned int* p, size_t n, unsigned int bits) {
    size_t i = (size_t)blockIdx.x * blockDim.x + threadIdx.x;
    size_t stride = (size_t)gridDim.x * blockDim.x;
    for (; i < n; i += stride) p[i] = bits;
}

// ---------------------------------------------------- atom encoder (gather)
__global__ void k_atom(const int* __restrict__ xi,
                       const float* t0, const float* t1, const float* t2,
                       const float* t3, const float* t4, const float* t5,
                       const float* t6, const float* t7, const float* t8,
                       float* __restrict__ feat, int N) {
    int wave = threadIdx.x >> 5, lane = threadIdx.x & 31;
    int n = blockIdx.x * 8 + wave;
    if (n >= N) return;
    int my = (lane < 9) ? xi[n * 9 + lane] : 0;
    const float* tabs[9] = { t0, t1, t2, t3, t4, t5, t6, t7, t8 };
    int f = lane * 4;
    float4 acc = make_float4(0.f, 0.f, 0.f, 0.f);
#pragma unroll
    for (int i = 0; i < 9; ++i) {
        int id = __shfl(my, i, 32);
        const float4 v = *(const float4*)(tabs[i] + (size_t)id * 128 + f);
        acc.x += v.x; acc.y += v.y; acc.z += v.z; acc.w += v.w;
    }
    *(float4*)(feat + (size_t)n * 128 + f) = acc;
}

// --------------------------------------- W (fp32, [K,N]) -> bf16 transposed
__global__ void k_convert_w(const float* __restrict__ W, unsigned short* __restrict__ WtBf) {
    int idx = blockIdx.x * 256 + threadIdx.x;   // 16384 elements
    int n = idx >> 7, k = idx & 127;
    union { __bf16 b; unsigned short u16; } cv;
    cv.b = f2bf(W[k * 128 + n]);
    WtBf[idx] = cv.u16;
}

// --------------------------- H = A(fp32->bf16) @ W  via v_wmma_f32_16x16x32_bf16
__global__ void k_gemm_bf16(const float* __restrict__ A,
                            const unsigned short* __restrict__ WtBf,  // [N=128][K=128] bf16
                            float* __restrict__ Hout, int N) {
    __shared__ unsigned int lwu[8192];                       // 128x128 bf16 = 32 KB
    const unsigned int* wsrc = (const unsigned int*)WtBf;
    int tid = threadIdx.x;
#pragma unroll
    for (int i = 0; i < 32; ++i) lwu[tid + i * 256] = wsrc[tid + i * 256];
    __syncthreads();
    const __bf16* lw = (const __bf16*)lwu;

    int wave = tid >> 5, lane = tid & 31;
    int m0 = blockIdx.x * 128 + wave * 16;
    if (m0 >= N) return;
    int lhalf = (lane >> 4) & 1;
    int kh = lhalf * 8;
    int l15 = lane & 15;
    int ra = m0 + l15; if (ra >= N) ra = N - 1;
    const float* arow = A + (size_t)ra * 128;

    v8f acc[8] = {};
#pragma unroll
    for (int k0 = 0; k0 < 128; k0 += 32) {
        float4 a0 = *(const float4*)(arow + k0 + kh);
        float4 a1 = *(const float4*)(arow + k0 + kh + 4);
        float4 a2 = *(const float4*)(arow + k0 + kh + 16);
        float4 a3 = *(const float4*)(arow + k0 + kh + 20);
        v16bf af;
        af[0]  = f2bf(a0.x); af[1]  = f2bf(a0.y); af[2]  = f2bf(a0.z); af[3]  = f2bf(a0.w);
        af[4]  = f2bf(a1.x); af[5]  = f2bf(a1.y); af[6]  = f2bf(a1.z); af[7]  = f2bf(a1.w);
        af[8]  = f2bf(a2.x); af[9]  = f2bf(a2.y); af[10] = f2bf(a2.z); af[11] = f2bf(a2.w);
        af[12] = f2bf(a3.x); af[13] = f2bf(a3.y); af[14] = f2bf(a3.z); af[15] = f2bf(a3.w);
#pragma unroll
        for (int nt = 0; nt < 8; ++nt) {
            const __bf16* bp = lw + ((size_t)(nt * 16 + l15) * 128) + k0 + kh;
            v16bf bf_;
#pragma unroll
            for (int e = 0; e < 8; ++e) { bf_[e] = bp[e]; bf_[e + 8] = bp[e + 16]; }
            acc[nt] = __builtin_amdgcn_wmma_f32_16x16x32_bf16(
                false, af, false, bf_, (short)0, acc[nt], false, false);
        }
    }
#pragma unroll
    for (int nt = 0; nt < 8; ++nt) {
        int c = nt * 16 + l15;
#pragma unroll
        for (int j = 0; j < 8; ++j) {
            int r = m0 + j + lhalf * 8;
            if (r < N) Hout[(size_t)r * 128 + c] = acc[nt][j];
        }
    }
}

// ------------------------------------------- per-node attention projections
__global__ void k_alpha(const float* __restrict__ Hm, const float* __restrict__ asrc,
                        const float* __restrict__ adst, float* __restrict__ als,
                        float* __restrict__ ald, int N) {
    int t = blockIdx.x * 256 + threadIdx.x;
    if (t >= 2 * N) return;
    int n = t >> 1, hd = t & 1;
    const float* hp = Hm + (size_t)n * 128 + hd * 64;
    const float* as_ = asrc + hd * 64;
    const float* ad_ = adst + hd * 64;
    float s = 0.f, d = 0.f;
#pragma unroll 8
    for (int c = 0; c < 64; ++c) { float v = hp[c]; s += v * as_[c]; d += v * ad_[c]; }
    als[t] = s; ald[t] = d;
}

__device__ __forceinline__ void edge_sd(const int* __restrict__ ei, long E, long j,
                                        int& s, int& d) {
    if (j < E) { s = ei[j]; d = ei[E + j]; }
    else       { s = d = (int)(j - E); }          // self-loops appended
}

// ------------------------------------------------ softmax pass 1: segment max
__global__ void k_edge_max(const int* __restrict__ ei, long E, long E2,
                           const float* __restrict__ als, const float* __restrict__ ald,
                           float* __restrict__ m) {
    long j = (long)blockIdx.x * 256 + threadIdx.x;
    if (j >= E2) return;
    int s, d; edge_sd(ei, E, j, s, d);
#pragma unroll
    for (int h = 0; h < 2; ++h) {
        float e = als[s * 2 + h] + ald[d * 2 + h];
        e = (e > 0.f) ? e : 0.2f * e;             // leaky_relu(0.2)
        atomicMaxF(&m[d * 2 + h], e);
    }
}

// -------------------------------------------- softmax pass 2: exp + segment sum
__global__ void k_edge_expsum(const int* __restrict__ ei, long E, long E2,
                              const float* __restrict__ als, const float* __restrict__ ald,
                              const float* __restrict__ m, float* __restrict__ den,
                              float* __restrict__ exb) {
    long j = (long)blockIdx.x * 256 + threadIdx.x;
    if (j >= E2) return;
    int s, d; edge_sd(ei, E, j, s, d);
#pragma unroll
    for (int h = 0; h < 2; ++h) {
        float e = als[s * 2 + h] + ald[d * 2 + h];
        e = (e > 0.f) ? e : 0.2f * e;
        float ex = __expf(e - m[d * 2 + h]);
        exb[j * 2 + h] = ex;
        atomicAdd(&den[d * 2 + h], ex);
    }
}

// --------------------------------------- message aggregation (wave per edge)
__global__ void k_edge_agg(const int* __restrict__ ei, long E, long E2,
                           const float* __restrict__ Hm, const float* __restrict__ exb,
                           const float* __restrict__ den, float* __restrict__ out) {
    long j = (long)blockIdx.x * 8 + (threadIdx.x >> 5);
    if (j >= E2) return;
    int lane = threadIdx.x & 31;
    int s, d; edge_sd(ei, E, j, s, d);
    int f = lane * 4;
    int head = f >> 6;
    float w = exb[j * 2 + head] / den[d * 2 + head];
    const float4 hv = *(const float4*)(Hm + (size_t)s * 128 + f);
    float* op = out + (size_t)d * 128 + f;
    atomicAdd(op + 0, w * hv.x);
    atomicAdd(op + 1, w * hv.y);
    atomicAdd(op + 2, w * hv.z);
    atomicAdd(op + 3, w * hv.w);
}

// --------------------------------------------------- out = relu(out + bias)
__global__ void k_finalize(float* __restrict__ x, const float* __restrict__ b, size_t n) {
    size_t i = (size_t)blockIdx.x * 256 + threadIdx.x;
    if (i >= n) return;
    float v = x[i] + b[i & 127];
    x[i] = (v > 0.f) ? v : 0.f;
}

// ------------------------------------------------------- global mean pooling
__global__ void k_pool(const float* __restrict__ x1, const float* __restrict__ x2,
                       const int* __restrict__ batch, float* __restrict__ pooled,
                       float* __restrict__ cnt, int N) {
    int wave = threadIdx.x >> 5, lane = threadIdx.x & 31;
    int n = blockIdx.x * 8 + wave;
    if (n >= N) return;
    int g = batch[n];
    int f = lane * 4;
    const float4 v1 = *(const float4*)(x1 + (size_t)n * 128 + f);
    const float4 v2 = *(const float4*)(x2 + (size_t)n * 128 + f);
    float* pp = pooled + (size_t)g * 128 + f;
    atomicAdd(pp + 0, v1.x + v2.x);
    atomicAdd(pp + 1, v1.y + v2.y);
    atomicAdd(pp + 2, v1.z + v2.z);
    atomicAdd(pp + 3, v1.w + v2.w);
    if (lane == 0) atomicAdd(&cnt[g], 2.0f);
}

// ---------------------------------------------------------- MLP head (tiny)
__global__ void k_mlp(const float* __restrict__ pooled, const float* __restrict__ cnt,
                      const float* __restrict__ lw1, const float* __restrict__ lb1,
                      const float* __restrict__ lw2, const float* __restrict__ lb2,
                      const float* __restrict__ lw3, const float* __restrict__ lb3,
                      float* __restrict__ out) {
    __shared__ float pr[128], h1[64], h2[32];
    int g = blockIdx.x, t = threadIdx.x;     // block = 64 threads
    float inv = 1.0f / fmaxf(cnt[g], 1.0f);
    pr[t]      = pooled[(size_t)g * 128 + t] * inv;
    pr[t + 64] = pooled[(size_t)g * 128 + 64 + t] * inv;
    __syncthreads();
    float s = lb1[t];
    for (int k = 0; k < 128; ++k) s += pr[k] * lw1[k * 64 + t];
    h1[t] = fmaxf(s, 0.f);
    __syncthreads();
    if (t < 32) {
        float s2 = lb2[t];
        for (int k = 0; k < 64; ++k) s2 += h1[k] * lw2[k * 32 + t];
        h2[t] = fmaxf(s2, 0.f);
    }
    __syncthreads();
    if (t < 2) {
        float s3 = lb3[t];
        for (int k = 0; k < 32; ++k) s3 += h2[k] * lw3[k * 2 + t];
        out[g * 2 + t] = s3;
    }
}

// --------------------------------------------------------------- host driver
static void run_gat_layer(const float* xin, const float* Wmat, const float* asrc,
                          const float* adst, const float* bias, float* hbuf,
                          unsigned short* wbf, float* als, float* ald, float* mbuf,
                          float* den, float* exb, float* xout, const int* ei,
                          int N, long E, long E2, hipStream_t stream) {
    k_convert_w<<<64, 256, 0, stream>>>(Wmat, wbf);
    k_gemm_bf16<<<(N + 127) / 128, 256, 0, stream>>>(xin, wbf, hbuf, N);
    k_alpha<<<(2 * N + 255) / 256, 256, 0, stream>>>(hbuf, asrc, adst, als, ald, N);
    k_fill<<<256, 256, 0, stream>>>((unsigned int*)mbuf, (size_t)N * 2, 0xFF800000u);
    k_fill<<<256, 256, 0, stream>>>((unsigned int*)den, (size_t)N * 2, 0u);
    k_fill<<<2048, 256, 0, stream>>>((unsigned int*)xout, (size_t)N * 128, 0u);
    k_edge_max<<<(int)((E2 + 255) / 256), 256, 0, stream>>>(ei, E, E2, als, ald, mbuf);
    k_edge_expsum<<<(int)((E2 + 255) / 256), 256, 0, stream>>>(ei, E, E2, als, ald, mbuf, den, exb);
    k_edge_agg<<<(int)((E2 + 7) / 8), 256, 0, stream>>>(ei, E, E2, hbuf, exb, den, xout);
    k_finalize<<<(int)(((size_t)N * 128 + 255) / 256), 256, 0, stream>>>(xout, bias, (size_t)N * 128);
}

extern "C" void kernel_launch(void* const* d_in, const int* in_sizes, int n_in,
                              void* d_out, int out_size, void* d_ws, size_t ws_size,
                              hipStream_t stream) {
    const int N  = in_sizes[0] / 9;
    const long E = in_sizes[1] / 2;
    const long E2 = E + N;

    const int*   x     = (const int*)d_in[0];
    const int*   ei    = (const int*)d_in[1];
    const int*   batch = (const int*)d_in[2];
    const float* tabs[9];
    for (int i = 0; i < 9; ++i) tabs[i] = (const float*)d_in[3 + i];
    const float* W1  = (const float*)d_in[12];
    const float* a1s = (const float*)d_in[13];
    const float* a1d = (const float*)d_in[14];
    const float* b1  = (const float*)d_in[15];
    const float* W2  = (const float*)d_in[16];
    const float* a2s = (const float*)d_in[17];
    const float* a2d = (const float*)d_in[18];
    const float* b2  = (const float*)d_in[19];
    const float* lw1 = (const float*)d_in[20];
    const float* lb1 = (const float*)d_in[21];
    const float* lw2 = (const float*)d_in[22];
    const float* lb2 = (const float*)d_in[23];
    const float* lw3 = (const float*)d_in[24];
    const float* lb3 = (const float*)d_in[25];
    float* out = (float*)d_out;

    // ---- workspace carve-up (256 B aligned)
    char* ws = (char*)d_ws;
    size_t off = 0;
    auto carve = [&](size_t bytes) -> char* {
        char* p = ws + off;
        off = (off + bytes + 255) & ~(size_t)255;
        return p;
    };
    const size_t nf = (size_t)N * 128;
    float*          feat   = (float*)carve(nf * 4);
    float*          hbuf   = (float*)carve(nf * 4);
    float*          x1     = (float*)carve(nf * 4);
    float*          x2     = (float*)carve(nf * 4);
    float*          als    = (float*)carve((size_t)N * 2 * 4);
    float*          ald    = (float*)carve((size_t)N * 2 * 4);
    float*          mbuf   = (float*)carve((size_t)N * 2 * 4);
    float*          den    = (float*)carve((size_t)N * 2 * 4);
    float*          exb    = (float*)carve((size_t)E2 * 2 * 4);
    unsigned short* wbf    = (unsigned short*)carve(128 * 128 * 2);
    float*          pooled = (float*)carve(128 * 128 * 4);
    float*          cnt    = (float*)carve(128 * 4);

    // ---- pipeline
    k_atom<<<(N + 7) / 8, 256, 0, stream>>>(x, tabs[0], tabs[1], tabs[2], tabs[3],
                                            tabs[4], tabs[5], tabs[6], tabs[7], tabs[8],
                                            feat, N);
    run_gat_layer(feat, W1, a1s, a1d, b1, hbuf, wbf, als, ald, mbuf, den, exb, x1,
                  ei, N, E, E2, stream);
    run_gat_layer(x1, W2, a2s, a2d, b2, hbuf, wbf, als, ald, mbuf, den, exb, x2,
                  ei, N, E, E2, stream);

    k_fill<<<64, 256, 0, stream>>>((unsigned int*)pooled, (size_t)128 * 128, 0u);
    k_fill<<<1, 128, 0, stream>>>((unsigned int*)cnt, 128, 0u);
    k_pool<<<(N + 7) / 8, 256, 0, stream>>>(x1, x2, batch, pooled, cnt, N);
    k_mlp<<<128, 64, 0, stream>>>(pooled, cnt, lw1, lb1, lw2, lb2, lw3, lb3, out);
}
